// Attention_24129126269281
// MI455X (gfx1250) — matrix operand order repaired and checked by reading, not compiled, not run on hardware
//
#include <hip/hip_runtime.h>
#include <hip/hip_bf16.h>

typedef __attribute__((ext_vector_type(16))) _Float16 v16h;
typedef __attribute__((ext_vector_type(8)))  float    v8f;

#define B_   4096
#define T_   200
#define D_   64
#define TP_  208                      // 13 tiles of 16 time steps
#define NEG_INF_ (-4294967295.0f)     // -(2^32)+1, matches reference

// -------------------------------------------------------------------------
// Workspace layout:
//   [0,    4096)  : float   w1ac[64*16]     = W1a + W1c   (query-side fold)
//   [4096, 8192)  : _Float16 bfrag[4][32][16] in WMMA B-matrix layout:
//                   frag 0: (W1b - W1c) K=0..31   frag 1: (W1b - W1c) K=32..63
//                   frag 2:  W1d        K=0..31   frag 3:  W1d        K=32..63
//   B 32x16 layout (16-bit): lanes 0-15 hold K=0..7,16..23 (half h -> K=h, h+8 for h>=8),
//                            lanes 16-31 hold K=8..15,24..31; N = lane&15.
// -------------------------------------------------------------------------
__global__ __launch_bounds__(256) void din_prep(const float* __restrict__ W1,
                                                void* __restrict__ ws)
{
  float*    w1ac = (float*)ws;
  _Float16* bf   = (_Float16*)((char*)ws + 4096);

  for (int idx = threadIdx.x; idx < 64 * 16; idx += blockDim.x) {
    int d = idx >> 4, nn = idx & 15;
    w1ac[idx] = W1[d * 16 + nn] + W1[(128 + d) * 16 + nn];
  }
  for (int idx = threadIdx.x; idx < 4 * 512; idx += blockDim.x) {
    int f = idx >> 9;            // fragment 0..3
    int r = idx & 511;
    int l = r >> 4;              // lane 0..31
    int h = r & 15;              // half within v16h
    int k = ((f & 1) << 5) + ((h >> 3) << 4) + ((l >> 4) << 3) + (h & 7);   // K index 0..63
    int nn = l & 15;
    float v;
    if (f < 2) v = W1[(64 + k) * 16 + nn] - W1[(128 + k) * 16 + nn];
    else       v = W1[(192 + k) * 16 + nn];
    bf[idx] = (_Float16)v;
  }
}

// -------------------------------------------------------------------------
// One wave32 per batch row.
// -------------------------------------------------------------------------
__global__ __launch_bounds__(32) void din_main(
    const float* __restrict__ queries, const float* __restrict__ keys,
    const int*   __restrict__ klen,
    const float* __restrict__ b1, const float* __restrict__ W2,
    const float* __restrict__ b2, const float* __restrict__ W3,
    const float* __restrict__ b3, const float* __restrict__ W4,
    const float* __restrict__ b4, const void* __restrict__ ws,
    float* __restrict__ out)
{
  const int b     = blockIdx.x;
  const int lane  = threadIdx.x;
  const int n     = lane & 15;            // hidden unit / column index
  const int c1    = (lane >> 4) << 3;     // A-frag K-chunk base: 0 or 8
  const int mbase = (lane >> 4) << 3;     // C-frag row base: 0 or 8

  __shared__ float s_x[16 * 16];
  __shared__ float s_sc[TP_];
  __shared__ float s_att[T_];
  __shared__ float s_out[D_];

  const float* q  = queries + (size_t)b * D_;
  const float* kb = keys    + (size_t)b * T_ * D_;
  const int   len = klen[b];

  const float*    w1ac = (const float*)ws;
  const _Float16* bfp  = (const _Float16*)((const char*)ws + 4096);
  const v16h Bbc0 = *(const v16h*)(bfp + 0 * 512 + lane * 16);
  const v16h Bbc1 = *(const v16h*)(bfp + 1 * 512 + lane * 16);
  const v16h Bd0  = *(const v16h*)(bfp + 2 * 512 + lane * 16);
  const v16h Bd1  = *(const v16h*)(bfp + 3 * 512 + lane * 16);

  // Folded query term: qb = b1[n] + sum_d q[d] * (W1a+W1c)[d][n]
  float qb = b1[n];
  #pragma unroll 8
  for (int d = 0; d < D_; ++d) qb += q[d] * w1ac[d * 16 + n];

  // Query values at this lane's A-fragment K positions (for q*k fragments).
  // A 16-bit 16x32 layout: halves 0..7 -> K = c1+0..7, halves 8..15 -> K = 16+c1+0..7
  float qs[32];
  #pragma unroll
  for (int e = 0; e < 8; ++e) {
    qs[e]      = q[c1 + e];
    qs[8 + e]  = q[16 + c1 + e];
    qs[16 + e] = q[32 + c1 + e];
    qs[24 + e] = q[48 + c1 + e];
  }

  // ---- Phase A: scores for all T, 16 time steps per WMMA tile -------------
  for (int t0 = 0; t0 < TP_; t0 += 16) {
    int trow = t0 + n;                    // row M = time step within tile
    if (trow > T_ - 1) trow = T_ - 1;     // clamp pad rows (masked later)

    const float4* kr = (const float4*)(kb + (size_t)trow * D_ + c1);
    float kv[32];
    #pragma unroll
    for (int c = 0; c < 4; ++c) {         // chunks at d = c*16 + c1
      float4 x0 = kr[c * 4];
      float4 x1 = kr[c * 4 + 1];
      kv[c * 8 + 0] = x0.x; kv[c * 8 + 1] = x0.y;
      kv[c * 8 + 2] = x0.z; kv[c * 8 + 3] = x0.w;
      kv[c * 8 + 4] = x1.x; kv[c * 8 + 5] = x1.y;
      kv[c * 8 + 6] = x1.z; kv[c * 8 + 7] = x1.w;
    }
    {
      int pr = trow + 16; if (pr > T_ - 1) pr = T_ - 1;
      __builtin_prefetch(kb + (size_t)pr * D_ + c1, 0, 1);
    }

    v16h ak0, ak1, aq0, aq1;              // k and q*k fragments (d 0..31 / 32..63)
    #pragma unroll
    for (int h = 0; h < 16; ++h) {
      ak0[h] = (_Float16)kv[h];
      ak1[h] = (_Float16)kv[16 + h];
      aq0[h] = (_Float16)(kv[h]      * qs[h]);
      aq1[h] = (_Float16)(kv[16 + h] * qs[16 + h]);
    }

    v8f c8;
    #pragma unroll
    for (int j = 0; j < 8; ++j) c8[j] = qb;   // C init = folded query term + b1

    c8 = __builtin_amdgcn_wmma_f32_16x16x32_f16(false, ak0, false, Bbc0, (short)0, c8, false, false);
    c8 = __builtin_amdgcn_wmma_f32_16x16x32_f16(false, ak1, false, Bbc1, (short)0, c8, false, false);
    c8 = __builtin_amdgcn_wmma_f32_16x16x32_f16(false, aq0, false, Bd0,  (short)0, c8, false, false);
    c8 = __builtin_amdgcn_wmma_f32_16x16x32_f16(false, aq1, false, Bd1,  (short)0, c8, false, false);

    #pragma unroll
    for (int j = 0; j < 8; ++j)
      s_x[(mbase + j) * 16 + n] = 1.0f / (1.0f + __expf(-c8[j]));
    __syncthreads();

    // Tiny layers 2 & 3: one time step per lane (lanes 0..15)
    if (lane < 16) {
      float xr[16];
      #pragma unroll
      for (int i = 0; i < 16; ++i) xr[i] = s_x[lane * 16 + i];
      float sc = b3[0];
      #pragma unroll
      for (int o = 0; o < 8; ++o) {
        float acc = b2[o];
        #pragma unroll
        for (int i = 0; i < 16; ++i) acc += xr[i] * W2[i * 8 + o];
        sc += (1.0f / (1.0f + __expf(-acc))) * W3[o];
      }
      int t = t0 + lane;
      sc = (t < len) ? sc : NEG_INF_;     // mask BEFORE scaling, as reference
      s_sc[t] = sc * 0.125f;              // 1/sqrt(64)
    }
    __syncthreads();
  }

  // ---- Softmax over valid T range (wave-level reductions) -----------------
  float mx = -3.0e38f;
  for (int i = lane; i < T_; i += 32) mx = fmaxf(mx, s_sc[i]);
  #pragma unroll
  for (int off = 16; off >= 1; off >>= 1) mx = fmaxf(mx, __shfl_xor(mx, off, 32));
  float se = 0.0f;
  for (int i = lane; i < T_; i += 32) se += __expf(s_sc[i] - mx);
  #pragma unroll
  for (int off = 16; off >= 1; off >>= 1) se += __shfl_xor(se, off, 32);
  float inv = 1.0f / se;
  for (int i = lane; i < T_; i += 32) s_att[i] = __expf(s_sc[i] - mx) * inv;
  __syncthreads();

  // ---- Weighted key sum (keys slice is L2-hot: 50KB, just touched) --------
  const int d0 = lane << 1;
  float r0 = 0.0f, r1 = 0.0f;
  for (int t = 0; t < T_; ++t) {
    float w = s_att[t];
    const float2 kv2 = *(const float2*)(kb + (size_t)t * D_ + d0);
    r0 += w * kv2.x;
    r1 += w * kv2.y;
  }
  s_out[d0]     = r0;
  s_out[d0 + 1] = r1;
  __syncthreads();

  // ---- Fused 64x64 output projection --------------------------------------
  float o0 = b4[d0], o1 = b4[d0 + 1];
  #pragma unroll 8
  for (int k = 0; k < D_; ++k) {
    float ov = s_out[k];
    o0 += ov * W4[k * D_ + d0];
    o1 += ov * W4[k * D_ + d0 + 1];
  }
  out[(size_t)b * D_ + d0]     = o0;
  out[(size_t)b * D_ + d0 + 1] = o1;
}

extern "C" void kernel_launch(void* const* d_in, const int* in_sizes, int n_in,
                              void* d_out, int out_size, void* d_ws, size_t ws_size,
                              hipStream_t stream)
{
  const float* queries = (const float*)d_in[0];
  const float* keys    = (const float*)d_in[1];
  const int*   klenp   = (const int*)d_in[2];
  const float* W1 = (const float*)d_in[3];
  const float* b1 = (const float*)d_in[4];
  const float* W2 = (const float*)d_in[5];
  const float* b2 = (const float*)d_in[6];
  const float* W3 = (const float*)d_in[7];
  const float* b3 = (const float*)d_in[8];
  const float* W4 = (const float*)d_in[9];
  const float* b4 = (const float*)d_in[10];
  float* out = (float*)d_out;

  din_prep<<<1, 256, 0, stream>>>(W1, d_ws);
  din_main<<<B_, 32, 0, stream>>>(queries, keys, klenp, b1, W2, b2, W3, b3,
                                  W4, b4, d_ws, out);
}
